// PoolingModule_81484119540192
// MI455X (gfx1250) — matrix operand order repaired and checked by reading, not compiled, not run on hardware
//
#include <hip/hip_runtime.h>
#include <math.h>

// ---------------------------------------------------------------------------
// Types / helpers
// ---------------------------------------------------------------------------
typedef __attribute__((ext_vector_type(16))) _Float16 v16h;
typedef __attribute__((ext_vector_type(8)))  _Float16 v8h;
typedef __attribute__((ext_vector_type(8)))  float    v8f;
typedef __attribute__((ext_vector_type(4)))  unsigned u32x4;
typedef __attribute__((ext_vector_type(8)))  int      i32x8;
typedef __attribute__((ext_vector_type(4)))  int      i32x4;

#define LN_EPS 1e-5f
#define USE_TDM 1

static __device__ __forceinline__ v16h joinv(v8h lo, v8h hi) {
  return __builtin_shufflevector(lo, hi, 0,1,2,3,4,5,6,7,8,9,10,11,12,13,14,15);
}

static __device__ __forceinline__ v8f wmma16(v16h a, v16h b, v8f c) {
  // D = A(16x32 f16) * B(32x16 f16) + C(16x16 f32)
  return __builtin_amdgcn_wmma_f32_16x16x32_f16(false, a, false, b, (short)0, c,
                                                false, false);
}

// LDS byte offset of a __shared__ object (flat shared addr low 32 bits).
static __device__ __forceinline__ unsigned lds_off(const void* p) {
  return (unsigned)(uintptr_t)p;
}

#if USE_TDM
// ---------------------------------------------------------------------------
// Tensor Data Mover: 2D tile (global -> LDS). Descriptor per CDNA5 ISA D#:
//  group0: [1:0]count=1 | [63:32]lds_addr | [120:64]global_addr | [127:126]=2
//  group1: [17:16]data_size(2=4B) | tensor_dim0/1 | tile_dim0/1 | dim0_stride
// Dimensions in 4-byte units. Issued by one wave; tracked with TENSORcnt.
// ---------------------------------------------------------------------------
static __device__ __forceinline__ void tdm_load_2d(
    unsigned ldsaddr, const void* gptr,
    unsigned tensor_d0, unsigned tensor_d1,
    unsigned tile_d0, unsigned tile_d1,
    unsigned long long stride0)
{
  const unsigned long long ga = (unsigned long long)(uintptr_t)gptr;
  u32x4 g0;
  g0[0] = 1u;                                            // count=1, user mode
  g0[1] = ldsaddr;
  g0[2] = (unsigned)(ga & 0xFFFFFFFFu);
  g0[3] = (unsigned)((ga >> 32) & 0x01FFFFFFu) | (2u << 30);   // type=2
  i32x8 g1;
  g1[0] = (int)(2u << 16);                               // data_size = 4B
  g1[1] = (int)((tensor_d0 & 0xFFFFu) << 16);            // barrier_addr=0
  g1[2] = (int)((tensor_d0 >> 16) | ((tensor_d1 & 0xFFFFu) << 16));
  g1[3] = (int)((tensor_d1 >> 16) | (tile_d0 << 16));
  g1[4] = (int)tile_d1;                                  // tile_dim2 = 0
  g1[5] = (int)(stride0 & 0xFFFFFFFFu);
  g1[6] = (int)((stride0 >> 32) & 0xFFFFu);              // dim1_stride = 0
  g1[7] = 0;
  const i32x4 gz = {0, 0, 0, 0};
#if defined(__clang_major__) && (__clang_major__ >= 23)
  const i32x8 gz8 = {0, 0, 0, 0, 0, 0, 0, 0};
  __builtin_amdgcn_tensor_load_to_lds(g0, g1, gz, gz, gz8, 0);
#else
  __builtin_amdgcn_tensor_load_to_lds(g0, g1, gz, gz, 0);
#endif
}
#endif

// ---------------------------------------------------------------------------
// LayerNorm: one block per row, D must be 4096 (16 elems / thread, 256 thr).
// ---------------------------------------------------------------------------
__global__ __launch_bounds__(256) void ln_kernel(
    const float* __restrict__ in, const float* __restrict__ w,
    const float* __restrict__ b, _Float16* __restrict__ out16,
    float* __restrict__ out32, int D)
{
  const int row = blockIdx.x;
  const int tid = threadIdx.x;
  const float* xr = in + (size_t)row * D;

  float lx[16];
  float s = 0.f;
#pragma unroll
  for (int j = 0; j < 16; ++j) { lx[j] = xr[tid + j * 256]; s += lx[j]; }
#pragma unroll
  for (int d = 1; d < 32; d <<= 1) s += __shfl_xor(s, d, 32);

  __shared__ float red[16];
  if ((tid & 31) == 0) red[tid >> 5] = s;
  __syncthreads();
  float tot = 0.f;
#pragma unroll
  for (int i = 0; i < 8; ++i) tot += red[i];
  const float invD = 1.0f / (float)D;
  const float mu = tot * invD;

  float vs = 0.f;
#pragma unroll
  for (int j = 0; j < 16; ++j) { float d0 = lx[j] - mu; vs += d0 * d0; }
#pragma unroll
  for (int d = 1; d < 32; d <<= 1) vs += __shfl_xor(vs, d, 32);
  if ((tid & 31) == 0) red[8 + (tid >> 5)] = vs;
  __syncthreads();
  float vtot = 0.f;
#pragma unroll
  for (int i = 0; i < 8; ++i) vtot += red[8 + i];
  const float rstd = rsqrtf(vtot * invD + LN_EPS);

  _Float16* o16 = out16 + (size_t)row * D;
  float*    o32 = out32 ? out32 + (size_t)row * D : nullptr;
#pragma unroll
  for (int j = 0; j < 16; ++j) {
    int i = tid + j * 256;
    float y = (lx[j] - mu) * rstd * w[i] + b[i];
    o16[i] = (_Float16)y;
    if (o32) o32[i] = y;
  }
}

// ---------------------------------------------------------------------------
// f32 (K,N) weight -> f16 transposed (N,K).  32x32 LDS tile, 256 threads.
// ---------------------------------------------------------------------------
__global__ __launch_bounds__(256) void wtrans_kernel(
    const float* __restrict__ W, _Float16* __restrict__ Wt, int D)
{
  __shared__ float tile[32][33];
  const int k0 = blockIdx.x * 32, n0 = blockIdx.y * 32;
  const int tx = threadIdx.x & 31, ty = threadIdx.x >> 5;
#pragma unroll
  for (int i = ty; i < 32; i += 8)
    tile[i][tx] = W[(size_t)(k0 + i) * D + n0 + tx];
  __syncthreads();
#pragma unroll
  for (int i = ty; i < 32; i += 8)
    Wt[(size_t)(n0 + i) * D + k0 + tx] = (_Float16)tile[tx][i];
}

// f16 (Rr,Cc) -> f16 transposed (Cc,Rr)
__global__ __launch_bounds__(256) void htrans_kernel(
    const _Float16* __restrict__ in, _Float16* __restrict__ out, int Rr, int Cc)
{
  __shared__ _Float16 tile[32][33];
  const int c0 = blockIdx.x * 32, r0 = blockIdx.y * 32;
  const int tx = threadIdx.x & 31, ty = threadIdx.x >> 5;
#pragma unroll
  for (int i = ty; i < 32; i += 8)
    tile[i][tx] = in[(size_t)(r0 + i) * Cc + c0 + tx];
  __syncthreads();
#pragma unroll
  for (int i = ty; i < 32; i += 8)
    out[(size_t)(c0 + i) * Rr + r0 + tx] = tile[tx][i];
}

// ---------------------------------------------------------------------------
// Tiled f16 GEMM: C(MxN) = A(MxK,row) * Bt(NxK,row)^T ; fp32 accumulate.
// Block tile 128x128, BK=32, 8 waves, each wave 32x64 via 2x4 WMMAs.
// Tiles staged into double-buffered LDS by the Tensor Data Mover (wave 0
// issues the D#; others sync via barriers after s_wait_tensorcnt).
// ---------------------------------------------------------------------------
enum { EPI_F32 = 0, EPI_F16 = 1, EPI_GELU = 2, EPI_ADD = 3 };

template <int EPI>
__global__ __launch_bounds__(256) void gemm_kernel(
    const _Float16* __restrict__ A, const _Float16* __restrict__ Bt,
    int M, int N, int K,
    float* __restrict__ C32, _Float16* __restrict__ C16,
    const float* __restrict__ res1, const float* __restrict__ res2)
{
  __shared__ alignas(128) _Float16 As[2][128 * 32];
  __shared__ alignas(128) _Float16 Bs[2][128 * 32];

  const int tid  = threadIdx.x;
  const int lane = tid & 31;
  const int wave = tid >> 5;
  const int wm   = wave >> 1;      // 0..3 (M direction)
  const int wn   = wave & 1;       // 0..1 (N direction)
  const int half = lane >> 4;      // 0/1
  const int lrow = lane & 15;

  const size_t bM = (size_t)blockIdx.y * 128;
  const size_t bN = (size_t)blockIdx.x * 128;
  const bool issuer = (wave == 0);

  const v8f z8 = {0.f, 0.f, 0.f, 0.f, 0.f, 0.f, 0.f, 0.f};
  v8f acc[2][4];
#pragma unroll
  for (int t = 0; t < 2; ++t)
#pragma unroll
    for (int u = 0; u < 4; ++u) acc[t][u] = z8;

  const unsigned Kdw = (unsigned)(K >> 1);   // row length in dwords

#if USE_TDM
  if (issuer) {
    tdm_load_2d(lds_off(&As[0][0]), A  + bM * K, Kdw, 128u, 16u, 128u, Kdw);
    tdm_load_2d(lds_off(&Bs[0][0]), Bt + bN * K, Kdw, 128u, 16u, 128u, Kdw);
    __builtin_amdgcn_s_wait_tensorcnt(0);
  }
  __syncthreads();
#endif

  int buf = 0;
  for (int k0 = 0; k0 < K; k0 += 32) {
#if USE_TDM
    if (issuer && (k0 + 32 < K)) {      // prefetch next tiles via TDM
      tdm_load_2d(lds_off(&As[buf ^ 1][0]), A  + bM * K + (k0 + 32),
                  Kdw, 128u, 16u, 128u, Kdw);
      tdm_load_2d(lds_off(&Bs[buf ^ 1][0]), Bt + bN * K + (k0 + 32),
                  Kdw, 128u, 16u, 128u, Kdw);
    }
#else
#pragma unroll
    for (int p = 0; p < 2; ++p) {
      const int c = tid + p * 256;
      const int r = c >> 2;
      const int col = (c & 3) * 8;
      *(v8h*)&As[0][r * 32 + col] = *(const v8h*)&A [(bM + r) * K + k0 + col];
      *(v8h*)&Bs[0][r * 32 + col] = *(const v8h*)&Bt[(bN + r) * K + k0 + col];
    }
    __syncthreads();
#endif

    v16h af[2], bf[4];
#pragma unroll
    for (int t = 0; t < 2; ++t) {
      const _Float16* pa =
          &As[buf][(wm * 32 + t * 16 + lrow) * 32 + half * 16];
      af[t] = *(const v16h*)pa;
    }
#pragma unroll
    for (int u = 0; u < 4; ++u) {
      const _Float16* pb = &Bs[buf][(wn * 64 + u * 16 + lrow) * 32 + half * 16];
      bf[u] = *(const v16h*)pb;
    }
#pragma unroll
    for (int t = 0; t < 2; ++t)
#pragma unroll
      for (int u = 0; u < 4; ++u) acc[t][u] = wmma16(af[t], bf[u], acc[t][u]);

    __syncthreads();                    // all waves done with current buffer
#if USE_TDM
    if (issuer) __builtin_amdgcn_s_wait_tensorcnt(0);
    __syncthreads();                    // next buffer now visible to all
    buf ^= 1;
#endif
  }

  // Epilogue. C layout: VGPR v -> row v (lanes 0-15) or v+8 (lanes 16-31),
  // col = lane&15.
#pragma unroll
  for (int t = 0; t < 2; ++t) {
    const size_t rbase = bM + wm * 32 + t * 16 + half * 8;
#pragma unroll
    for (int u = 0; u < 4; ++u) {
      const size_t col = bN + wn * 64 + u * 16 + lrow;
#pragma unroll
      for (int v = 0; v < 8; ++v) {
        const size_t idx = (rbase + v) * (size_t)N + col;
        float val = acc[t][u][v];
        if (EPI == EPI_F32) {
          C32[idx] = val;
        } else if (EPI == EPI_F16) {
          C16[idx] = (_Float16)val;
        } else if (EPI == EPI_GELU) {
          C16[idx] = (_Float16)(0.5f * val * (1.0f + erff(val * 0.70710678118654752f)));
        } else { // EPI_ADD
          val += res1[idx];
          if (res2) val += res2[idx];
          C32[idx] = val;
        }
      }
    }
  }
}

// ---------------------------------------------------------------------------
// Fused cross-attention: block = 128 tokens x 1 head, 8 waves, 16 rows/wave.
// scores = q @ k^T (rows of k used directly as Bt), online softmax,
// O += P @ V via vt (pre-transposed V), P relayout through per-wave LDS.
// k/v are tiny (4 MB) -> L2-resident; fragments loaded directly from global
// with hoisted base pointers (32-bit inner offsets).
// ---------------------------------------------------------------------------
__global__ __launch_bounds__(256) void attn_kernel(
    const _Float16* __restrict__ q, const _Float16* __restrict__ k,
    const _Float16* __restrict__ vt, _Float16* __restrict__ o,
    int T, int R, int D)
{
  const int HD = 128;
  __shared__ alignas(64) _Float16 Pbuf[8][16 * 128];   // 32 KB

  const int tid  = threadIdx.x;
  const int lane = tid & 31;
  const int wave = tid >> 5;
  const int half = lane >> 4;
  const int lrow = lane & 15;
  const int head = blockIdx.y;
  const size_t rowbase = (size_t)blockIdx.x * 128 + wave * 16;
  const float scale = 0.08838834764831845f;            // 1/sqrt(128)

  // Hoisted 64-bit bases; inner-loop offsets are small ints.
  const _Float16* qbase =
      q + (rowbase + lrow) * (size_t)D + head * HD + half * 16;
  const _Float16* kbase = k + (size_t)lrow * D + head * HD + half * 16;
  const _Float16* vbase = vt + (size_t)(head * HD + lrow) * R + half * 16;
  _Float16* obase = o + (rowbase + half * 8) * (size_t)D + head * HD + lrow;

  const v8f z8 = {0.f, 0.f, 0.f, 0.f, 0.f, 0.f, 0.f, 0.f};
  v8f oacc[8];
#pragma unroll
  for (int u = 0; u < 8; ++u) oacc[u] = z8;
  float Mrow[8], Lrow[8];
#pragma unroll
  for (int v = 0; v < 8; ++v) { Mrow[v] = -INFINITY; Lrow[v] = 0.f; }

  _Float16* Pw = &Pbuf[wave][0];

  for (int rt = 0; rt < R; rt += 128) {
    // ---- scores tile: 16 x 128, K = HD = 128 --------------------------
    v8f sacc[8];
#pragma unroll
    for (int u = 0; u < 8; ++u) sacc[u] = z8;
#pragma unroll
    for (int kk = 0; kk < 128; kk += 32) {
      const _Float16* pa = qbase + kk;
      v16h af = *(const v16h*)pa;
#pragma unroll
      for (int u = 0; u < 8; ++u) {
        const _Float16* pb = kbase + (rt + u * 16) * D + kk;
        sacc[u] = wmma16(af, *(const v16h*)pb, sacc[u]);
      }
    }

    // ---- online softmax per row (reductions within 16-lane half-group) --
#pragma unroll
    for (int v = 0; v < 8; ++v) {
      float m = -INFINITY;
#pragma unroll
      for (int u = 0; u < 8; ++u) m = fmaxf(m, sacc[u][v]);
      m *= scale;
#pragma unroll
      for (int d = 1; d < 16; d <<= 1) m = fmaxf(m, __shfl_xor(m, d, 32));
      const float newM  = fmaxf(Mrow[v], m);
      const float alpha = __expf(Mrow[v] - newM);
      float lsum = 0.f;
#pragma unroll
      for (int u = 0; u < 8; ++u) {
        float pv = __expf(sacc[u][v] * scale - newM);
        lsum += pv;
        Pw[(v + half * 8) * 128 + u * 16 + lrow] = (_Float16)pv;
      }
#pragma unroll
      for (int d = 1; d < 16; d <<= 1) lsum += __shfl_xor(lsum, d, 32);
      Lrow[v] = Lrow[v] * alpha + lsum;
      Mrow[v] = newM;
#pragma unroll
      for (int ou = 0; ou < 8; ++ou) oacc[ou][v] *= alpha;
    }

    // Make the per-wave LDS P-tile visible to this wave's ds_loads.
    asm volatile("s_wait_dscnt 0" ::: "memory");

    // ---- O += P(16x128) @ V(128x128) via vt (Bt rows contiguous) -------
#pragma unroll
    for (int kk = 0; kk < 128; kk += 32) {
      const _Float16* pp = &Pw[lrow * 128 + kk + half * 16];
      v16h af = *(const v16h*)pp;
#pragma unroll
      for (int ou = 0; ou < 8; ++ou) {
        const _Float16* pb = vbase + (ou * 16) * R + rt + kk;
        oacc[ou] = wmma16(af, *(const v16h*)pb, oacc[ou]);
      }
    }
  }

  // ---- normalize + store f16 output ----------------------------------
#pragma unroll
  for (int v = 0; v < 8; ++v) {
    const float inv = 1.0f / Lrow[v];
#pragma unroll
    for (int ou = 0; ou < 8; ++ou)
      obase[v * D + ou * 16] = (_Float16)(oacc[ou][v] * inv);
  }
}

// ---------------------------------------------------------------------------
// Host orchestration
// ---------------------------------------------------------------------------
extern "C" void kernel_launch(void* const* d_in, const int* in_sizes, int n_in,
                              void* d_out, int out_size, void* d_ws,
                              size_t ws_size, hipStream_t stream)
{
  (void)in_sizes; (void)n_in; (void)out_size; (void)ws_size;
  const int D = 4096, R = 512, T = 8192, H = 32;

  const float* x       = (const float*)d_in[0];
  /* d_in[1] = seqlens: block-diagonal mask with identical latents == dense */
  const float* latents = (const float*)d_in[2];
  const float* ln_q_w  = (const float*)d_in[3];
  const float* ln_q_b  = (const float*)d_in[4];
  const float* ln_c_w  = (const float*)d_in[5];
  const float* ln_c_b  = (const float*)d_in[6];
  const float* Wq      = (const float*)d_in[7];
  const float* Wk      = (const float*)d_in[8];
  const float* Wv      = (const float*)d_in[9];
  const float* Wo      = (const float*)d_in[10];
  const float* ln_m_w  = (const float*)d_in[11];
  const float* ln_m_b  = (const float*)d_in[12];
  const float* W1      = (const float*)d_in[13];
  const float* W2      = (const float*)d_in[14];
  float* out = (float*)d_out;

  char* ws = (char*)d_ws;
  size_t off = 0;
  auto alloc = [&](size_t bytes) -> void* {
    void* p = ws + off;
    off = (off + bytes + 255) & ~(size_t)255;
    return p;
  };
  _Float16* S1  = (_Float16*)alloc((size_t)T * D * 2); // xq -> o -> gelu(t)
  _Float16* S2  = (_Float16*)alloc((size_t)T * D * 2); // q  -> y(f16)
  _Float16* Wt  = (_Float16*)alloc((size_t)D * D * 2); // shared W^T f16
  _Float16* xc  = (_Float16*)alloc((size_t)R * D * 2);
  _Float16* kb  = (_Float16*)alloc((size_t)R * D * 2);
  _Float16* vb  = (_Float16*)alloc((size_t)R * D * 2);
  _Float16* vtb = (_Float16*)alloc((size_t)D * R * 2);
  float*    y32 = (float*)   alloc((size_t)T * D * 4);

  const dim3 blk(256);
  const dim3 gW(D / 32, D / 32);
  const dim3 gBig(D / 128, T / 128);   // 32 x 64
  const dim3 gSm (D / 128, R / 128);   // 32 x 4

  // 1) layernorms
  ln_kernel<<<T, blk, 0, stream>>>(x, ln_q_w, ln_q_b, S1, nullptr, D);
  ln_kernel<<<R, blk, 0, stream>>>(latents, ln_c_w, ln_c_b, xc, nullptr, D);

  // 2) q/k/v projections (f16 out)
  wtrans_kernel<<<gW, blk, 0, stream>>>(Wq, Wt, D);
  gemm_kernel<EPI_F16><<<gBig, blk, 0, stream>>>(S1, Wt, T, D, D,
                                                 nullptr, S2, nullptr, nullptr);
  wtrans_kernel<<<gW, blk, 0, stream>>>(Wk, Wt, D);
  gemm_kernel<EPI_F16><<<gSm, blk, 0, stream>>>(xc, Wt, R, D, D,
                                                nullptr, kb, nullptr, nullptr);
  wtrans_kernel<<<gW, blk, 0, stream>>>(Wv, Wt, D);
  gemm_kernel<EPI_F16><<<gSm, blk, 0, stream>>>(xc, Wt, R, D, D,
                                                nullptr, vb, nullptr, nullptr);
  htrans_kernel<<<dim3(D / 32, R / 32), blk, 0, stream>>>(vb, vtb, R, D);

  // 3) fused attention: o = softmax(q k^T / sqrt(hd)) v   (S1 now holds o)
  attn_kernel<<<dim3(T / 128, H), blk, 0, stream>>>(S2, kb, vtb, S1, T, R, D);

  // 4) hiddens = o @ Wo + x  -> d_out (fp32)
  wtrans_kernel<<<gW, blk, 0, stream>>>(Wo, Wt, D);
  gemm_kernel<EPI_ADD><<<gBig, blk, 0, stream>>>(S1, Wt, T, D, D,
                                                 out, nullptr, x, nullptr);

  // 5) y = LN(hiddens), kept in both f16 (S2) and f32 (y32)
  ln_kernel<<<T, blk, 0, stream>>>(out, ln_m_w, ln_m_b, S2, y32, D);

  // 6) t = gelu(y @ W1) f16 -> S1
  wtrans_kernel<<<gW, blk, 0, stream>>>(W1, Wt, D);
  gemm_kernel<EPI_GELU><<<gBig, blk, 0, stream>>>(S2, Wt, T, D, D,
                                                  nullptr, S1, nullptr, nullptr);

  // 7) out = t @ W2 + y + hiddens   (reads d_out as residual, rewrites it)
  wtrans_kernel<<<gW, blk, 0, stream>>>(W2, Wt, D);
  gemm_kernel<EPI_ADD><<<gBig, blk, 0, stream>>>(S1, Wt, T, D, D,
                                                 out, nullptr, y32, out);
}
